// SAModule_4698694222638
// MI455X (gfx1250) — matrix operand order repaired and checked by reading, not compiled, not run on hardware
//
#include <hip/hip_runtime.h>
#include <hip/hip_bf16.h>

#define B_ 32
#define C_ 64
#define HEADS_ 4
#define D_ 16
#define N_ 1024
#define W_ 32
#define H_ 32

typedef __attribute__((ext_vector_type(16))) _Float16 v16h;
typedef __attribute__((ext_vector_type(8)))  _Float16 v8h;
typedef __attribute__((ext_vector_type(8)))  float    v8f;

#define LOG2E_ 1.4426950408889634f

__device__ __forceinline__ float sigmoidf_(float x) { return 1.0f / (1.0f + __expf(-x)); }

// ---------------------------------------------------------------------------
// Positional table (pre-scaled by log2(e) so scores land in exp2 domain):
// pt[h][n][d] (f16) = log2e * (rel_h[h,d,hh] + rel_w[h,d,ww]), n = ww*32+hh
// ---------------------------------------------------------------------------
__global__ void pos_kernel(const float* __restrict__ rel_h, const float* __restrict__ rel_w,
                           _Float16* __restrict__ pt) {
  int t = blockIdx.x * blockDim.x + threadIdx.x;     // h*1024 + n
  if (t >= HEADS_ * N_) return;
  int h = t >> 10, n = t & 1023;
  int ww = n >> 5, hh = n & 31;
  _Float16 row[16];
#pragma unroll
  for (int d = 0; d < D_; ++d)
    row[d] = (_Float16)((rel_h[(h * D_ + d) * H_ + hh] + rel_w[(h * D_ + d) * W_ + ww]) * LOG2E_);
  v8h r0, r1;
#pragma unroll
  for (int e = 0; e < 8; ++e) { r0[e] = row[e]; r1[e] = row[e + 8]; }
  v8h* dst = (v8h*)(pt + (size_t)t * D_);
  dst[0] = r0; dst[1] = r1;
}

// ---------------------------------------------------------------------------
// QKV projection on the matrix pipe: D[M=o, N=n] = W[o,c] x X[c,n].
// Block = (b, 128-wide n slab); 8 waves, one 16-wide n-tile each.
// Weights staged in LDS as f16 [o][c] (A reads = 2x ds_load_b128, contiguous);
// x slab staged [c][128] f16. 2 WMMAs per o-tile x 4 o-tiles x 3 matrices.
// kt is pre-scaled by log2(e) (see pos_kernel comment). Bias rides in C init.
// qt/kt out: [b,h,n,d] f16 ; vd out: [b,h,d,n] f16.
// ---------------------------------------------------------------------------
__global__ __launch_bounds__(256) void qkv_wmma_kernel(
    const float* __restrict__ x,
    const float* __restrict__ Wq, const float* __restrict__ bq,
    const float* __restrict__ Wk, const float* __restrict__ bk,
    const float* __restrict__ Wv, const float* __restrict__ bv,
    _Float16* __restrict__ qt, _Float16* __restrict__ kt, _Float16* __restrict__ vd)
{
  __shared__ _Float16 wsl[3 * C_ * C_];   // wq | wk*log2e | wv, 24KB
  __shared__ _Float16 xs[C_ * 128];       // x slab [c][128], 16KB
  const int b = blockIdx.x >> 3;
  const int nblk = blockIdx.x & 7;
  const int n0 = nblk * 128;

  for (int i = threadIdx.x; i < C_ * C_; i += 256) {
    wsl[i]               = (_Float16)Wq[i];
    wsl[C_ * C_ + i]     = (_Float16)(Wk[i] * LOG2E_);
    wsl[2 * C_ * C_ + i] = (_Float16)Wv[i];
  }
  for (int i = threadIdx.x; i < C_ * 128; i += 256) {
    int c = i >> 7, nn = i & 127;
    xs[i] = (_Float16)x[((size_t)(b * C_ + c)) * N_ + n0 + nn];
  }
  __syncthreads();

  const int wave = threadIdx.x >> 5;
  const int lane = threadIdx.x & 31;
  const int lo = lane & 15, hfl = lane >> 4;
  const int nloc = wave * 16 + lo;        // N position within slab
  const int gn = n0 + nloc;               // global n

  // B operands (shared across all o-tiles and all three matrices):
  // B[K=c, N=n]; halves e -> c = kc + hfl*8 + e (e<8), kc + 16 + hfl*8 + (e-8) (e>=8)
  v16h BX0, BX1;
#pragma unroll
  for (int e = 0; e < 16; ++e) {
    BX0[e] = xs[(hfl * 8 + (e & 8) + e) * 128 + nloc];
    BX1[e] = xs[(32 + hfl * 8 + (e & 8) + e) * 128 + nloc];
  }

  for (int m = 0; m < 3; ++m) {
    const _Float16* wmat = wsl + m * C_ * C_;
    const float* bias = (m == 0) ? bq : (m == 1) ? bk : bv;
    for (int ot = 0; ot < 4; ++ot) {      // o-tile == head index for q/k
      const _Float16* arow = wmat + (size_t)(ot * 16 + lo) * C_;
      v8h a00 = *(const v8h*)(arow + hfl * 8);
      v8h a01 = *(const v8h*)(arow + 16 + hfl * 8);
      v8h a10 = *(const v8h*)(arow + 32 + hfl * 8);
      v8h a11 = *(const v8h*)(arow + 48 + hfl * 8);
      v16h A0, A1;
#pragma unroll
      for (int e = 0; e < 8; ++e) { A0[e] = a00[e]; A0[e + 8] = a01[e]; A1[e] = a10[e]; A1[e + 8] = a11[e]; }

      v8f Dd;
#pragma unroll
      for (int r = 0; r < 8; ++r) Dd[r] = bias[ot * 16 + r + 8 * hfl];
      Dd = __builtin_amdgcn_wmma_f32_16x16x32_f16(false, A0, false, BX0, (short)0, Dd, false, false);
      Dd = __builtin_amdgcn_wmma_f32_16x16x32_f16(false, A1, false, BX1, (short)0, Dd, false, false);

      if (m < 2) {
        // D (r,lane) = (o = ot*16 + hfl*8 + r, n = gn): d-run of 8 is contiguous
        _Float16* dst = (m == 0) ? qt : kt;
        v8h rowv;
#pragma unroll
        for (int r = 0; r < 8; ++r) rowv[r] = (_Float16)Dd[r];
        *(v8h*)(dst + ((size_t)(b * HEADS_ + ot) * N_ + gn) * D_ + hfl * 8) = rowv;
      } else {
#pragma unroll
        for (int r = 0; r < 8; ++r) {
          int o = ot * 16 + r + 8 * hfl;
          vd[((size_t)(b * C_ + o)) * N_ + gn] = (_Float16)Dd[r];
        }
      }
    }
  }
}

// ---------------------------------------------------------------------------
// Fused attention. Reduction-free softmax:
//   - kt/pt pre-scaled by log2(e) => p = exp2(S), a bare v_exp_f32
//   - scores well-bounded (|S| << 100) so no max-shift needed in fp32
//   - row sums on the matrix pipe: L += P x Ones (one extra WMMA)
// Block = (b,h, quarter). 8 waves; each wave owns 2 i-tiles of 16 rows.
// Per 32-wide j-chunk: 4x v_wmma_f32_16x16x32_f16, zero cross-lane ops.
// ---------------------------------------------------------------------------
__global__ __launch_bounds__(256) void attn_kernel(
    const _Float16* __restrict__ qt, const _Float16* __restrict__ kt,
    const _Float16* __restrict__ vd, const _Float16* __restrict__ pt,
    float* __restrict__ att_out)
{
  __shared__ _Float16 pbuf[8 * 16 * 32];             // 1KB per wave
  const int bh   = blockIdx.x >> 2;                  // 0..127
  const int quad = blockIdx.x & 3;
  const int b = bh >> 2, h = bh & 3;
  const int wave = threadIdx.x >> 5;
  const int lane = threadIdx.x & 31;
  const int lo = lane & 15, hfl = lane >> 4;
  _Float16* pw = pbuf + wave * 512;

  const _Float16* qbase = qt + (size_t)bh * N_ * D_;
  const _Float16* kbase = kt + (size_t)bh * N_ * D_;
  const _Float16* vbase = vd + (size_t)bh * D_ * N_;
  const _Float16* pbase = pt + (size_t)h * N_ * D_;

  v16h BONES;
#pragma unroll
  for (int e = 0; e < 16; ++e) BONES[e] = (_Float16)1.0f;

  for (int t = 0; t < 2; ++t) {
    const int itile = quad * 16 + (t << 3) + wave;   // 0..63
    const int i0 = itile * 16 + lo;                  // this lane's A row (M = lo)

    // A operand: halves 0-7 = q[d = hfl*8 + e, i0], halves 8-15 = pos[d = hfl*8 + e, i0]
    v8h aq = *(const v8h*)(qbase + (size_t)i0 * D_ + hfl * 8);
    v8h apos = *(const v8h*)(pbase + (size_t)i0 * D_ + hfl * 8);
    v16h A;
#pragma unroll
    for (int e = 0; e < 8; ++e) { A[e] = aq[e]; A[e + 8] = apos[e]; }

    v8f O = {};   // output accumulator [i, d]
    v8f L = {};   // row-sum accumulator (all columns identical)

    for (int jc = 0; jc < N_; jc += 32) {
      // B operands for scores: halves 0-7 = k[d = hfl*8 + e, j], halves 8-15 = q[d = hfl*8 + e, j]
      const _Float16* bsrc = hfl ? qbase : kbase;
      int j0 = jc + lo, j1 = jc + 16 + lo;
      v8h t00 = *(const v8h*)(kbase + (size_t)j0 * D_ + hfl * 8);
      v8h t01 = *(const v8h*)(qbase + (size_t)j0 * D_ + hfl * 8);
      v8h t10 = *(const v8h*)(kbase + (size_t)j1 * D_ + hfl * 8);
      v8h t11 = *(const v8h*)(qbase + (size_t)j1 * D_ + hfl * 8);
      v16h B0, B1;
#pragma unroll
      for (int e = 0; e < 8; ++e) { B0[e] = t00[e]; B0[e + 8] = t01[e]; B1[e] = t10[e]; B1[e + 8] = t11[e]; }

      v8f S0 = {}, S1 = {};
      S0 = __builtin_amdgcn_wmma_f32_16x16x32_f16(false, A, false, B0, (short)0, S0, false, false);
      S1 = __builtin_amdgcn_wmma_f32_16x16x32_f16(false, A, false, B1, (short)0, S1, false, false);

      // B operand for O update: V^T[jc+K, d] = vd[d, jc+K]; lane = d, half selects K range
      v8h w0 = *(const v8h*)(vbase + (size_t)lo * N_ + jc + hfl * 8);
      v8h w1 = *(const v8h*)(vbase + (size_t)lo * N_ + jc + 16 + hfl * 8);
      v16h BV;
#pragma unroll
      for (int e = 0; e < 8; ++e) { BV[e] = w0[e]; BV[e + 8] = w1[e]; }

      // Unnormalized probabilities; scores already in log2 domain -> bare v_exp_f32.
      // D-layout: VGPR r -> row i = itile*16 + r + 8*hfl, lane lo -> col j.
      float p0[8], p1[8];
#pragma unroll
      for (int r = 0; r < 8; ++r) { p0[r] = exp2f(S0[r]); p1[r] = exp2f(S1[r]); }

      // C-layout -> A-layout transpose of P via per-wave private LDS bounce
#pragma unroll
      for (int r = 0; r < 8; ++r) {
        int irow = r + (hfl << 3);
        pw[irow * 32 + lo]      = (_Float16)p0[r];
        pw[irow * 32 + 16 + lo] = (_Float16)p1[r];
      }
      asm volatile("s_wait_dscnt 0" ::: "memory");
      v8h plo = *(const v8h*)(pw + lo * 32 + hfl * 8);
      v8h phi = *(const v8h*)(pw + lo * 32 + 16 + hfl * 8);
      v16h AP;
#pragma unroll
      for (int e = 0; e < 8; ++e) { AP[e] = plo[e]; AP[e + 8] = phi[e]; }

      O = __builtin_amdgcn_wmma_f32_16x16x32_f16(false, AP, false, BV, (short)0, O, false, false);
      L = __builtin_amdgcn_wmma_f32_16x16x32_f16(false, AP, false, BONES, (short)0, L, false, false);
    }

    // Normalize and store: out[b, c = h*16 + d, n = i]; lane = d, VGPR r -> i
#pragma unroll
    for (int r = 0; r < 8; ++r) {
      int i = itile * 16 + r + (hfl << 3);
      int c = h * D_ + lo;
      att_out[((size_t)(b * C_ + c)) * N_ + i] = O[r] / L[r];
    }
  }
}

// ---------------------------------------------------------------------------
// LayerNorm stats over (C,W,H) per batch.
// ---------------------------------------------------------------------------
__global__ __launch_bounds__(256) void lnstat_kernel(const float* __restrict__ att,
                                                     float* __restrict__ mu, float* __restrict__ rstd) {
  __shared__ float ssum[256], ssq[256];
  int b = blockIdx.x;
  const float* p = att + (size_t)b * C_ * N_;
  float s = 0.f, q = 0.f;
  for (int i = threadIdx.x; i < C_ * N_; i += 256) { float v = p[i]; s += v; q += v * v; }
  ssum[threadIdx.x] = s; ssq[threadIdx.x] = q;
  __syncthreads();
  for (int off = 128; off > 0; off >>= 1) {
    if (threadIdx.x < off) { ssum[threadIdx.x] += ssum[threadIdx.x + off]; ssq[threadIdx.x] += ssq[threadIdx.x + off]; }
    __syncthreads();
  }
  if (threadIdx.x == 0) {
    float mm = ssum[0] / (float)(C_ * N_);
    float vv = ssq[0] / (float)(C_ * N_) - mm * mm;
    mu[b] = mm; rstd[b] = rsqrtf(vv + 1e-5f);
  }
}

// ---------------------------------------------------------------------------
// CBAM channel pooling: avg/max over spatial per (b,c).
// ---------------------------------------------------------------------------
__global__ __launch_bounds__(256) void pool_kernel(const float* __restrict__ x,
                                                   float* __restrict__ avg, float* __restrict__ mxo) {
  __shared__ float sa[256], sm[256];
  int bc = blockIdx.x;
  const float* p = x + (size_t)bc * N_;
  float s = 0.f, m = -1e30f;
  for (int i = threadIdx.x; i < N_; i += 256) { float v = p[i]; s += v; m = fmaxf(m, v); }
  sa[threadIdx.x] = s; sm[threadIdx.x] = m;
  __syncthreads();
  for (int off = 128; off > 0; off >>= 1) {
    if (threadIdx.x < off) { sa[threadIdx.x] += sa[threadIdx.x + off]; sm[threadIdx.x] = fmaxf(sm[threadIdx.x], sm[threadIdx.x + off]); }
    __syncthreads();
  }
  if (threadIdx.x == 0) { avg[bc] = sa[0] / (float)N_; mxo[bc] = sm[0]; }
}

// ---------------------------------------------------------------------------
// CBAM channel MLP: ch_att = sigmoid(mlp(avg) + mlp(max)). Block per b, 64 threads.
// ---------------------------------------------------------------------------
__global__ void mlp_kernel(const float* __restrict__ avg, const float* __restrict__ mx,
                           const float* __restrict__ w1, const float* __restrict__ b1,
                           const float* __restrict__ w2, const float* __restrict__ b2,
                           float* __restrict__ ch) {
  __shared__ float ha[4], hm[4];
  int b = blockIdx.x, t = threadIdx.x;
  if (t < 4) {
    float sa_ = b1[t], sm_ = b1[t];
    for (int c = 0; c < C_; ++c) { sa_ += w1[t * C_ + c] * avg[b * C_ + c]; sm_ += w1[t * C_ + c] * mx[b * C_ + c]; }
    ha[t] = fmaxf(sa_, 0.f); hm[t] = fmaxf(sm_, 0.f);
  }
  __syncthreads();
  float oa = b2[t], om = b2[t];
  for (int j = 0; j < 4; ++j) { oa += w2[t * 4 + j] * ha[j]; om += w2[t * 4 + j] * hm[j]; }
  ch[b * C_ + t] = sigmoidf_(oa + om);
}

// ---------------------------------------------------------------------------
// Spatial feat: mean/max over channels of (ch_att * x).
// ---------------------------------------------------------------------------
__global__ void feat_kernel(const float* __restrict__ x, const float* __restrict__ ch,
                            float* __restrict__ fm, float* __restrict__ fx) {
  int t = blockIdx.x * blockDim.x + threadIdx.x;     // b*N + n
  int b = t >> 10;
  int n = t & 1023;
  float s = 0.f, m = -1e30f;
  for (int c = 0; c < C_; ++c) {
    float v = ch[b * C_ + c] * x[((size_t)(b * C_ + c)) * N_ + n];
    s += v; m = fmaxf(m, v);
  }
  fm[t] = s / (float)C_; fx[t] = m;
}

// ---------------------------------------------------------------------------
// 7x7 SAME conv over [mean;max] feat -> sigmoid -> spat[b,n].
// ---------------------------------------------------------------------------
__global__ void sconv_kernel(const float* __restrict__ fm, const float* __restrict__ fx,
                             const float* __restrict__ sw, const float* __restrict__ sb,
                             float* __restrict__ spat) {
  int t = blockIdx.x * blockDim.x + threadIdx.x;     // b*N + n
  int b = t >> 10, n = t & 1023;
  int w0 = n >> 5, h0 = n & 31;
  float acc = sb[0];
  for (int ci = 0; ci < 2; ++ci) {
    const float* f = ci ? fx : fm;
    for (int a = 0; a < 7; ++a) {
      int wi = w0 + a - 3;
      if (wi < 0 || wi >= W_) continue;
      for (int c2 = 0; c2 < 7; ++c2) {
        int hi2 = h0 + c2 - 3;
        if (hi2 < 0 || hi2 >= H_) continue;
        acc += f[b * N_ + wi * H_ + hi2] * sw[(ci * 7 + a) * 7 + c2];
      }
    }
  }
  spat[t] = sigmoidf_(acc);
}

// ---------------------------------------------------------------------------
// Final fuse: out = LN(att) + 2*x + spat * (ch_att * x)
// ---------------------------------------------------------------------------
__global__ __launch_bounds__(256) void final_kernel(
    const float* __restrict__ att, const float* __restrict__ x,
    const float* __restrict__ mu, const float* __restrict__ rstd,
    const float* __restrict__ g, const float* __restrict__ bb,
    const float* __restrict__ ch, const float* __restrict__ spat,
    float* __restrict__ out) {
  size_t t = (size_t)blockIdx.x * blockDim.x + threadIdx.x;
  int b = (int)(t >> 16);                            // C_*N_ = 65536
  int cn = (int)(t & 65535);
  int c = cn >> 10, n = cn & 1023;
  float a = att[t];
  float ln = (a - mu[b]) * rstd[b] * g[cn] + bb[cn];
  float xv = x[t];
  float xc = ch[b * C_ + c] * xv;
  out[t] = ln + 2.0f * xv + spat[b * N_ + n] * xc;
}

// ---------------------------------------------------------------------------
extern "C" void kernel_launch(void* const* d_in, const int* in_sizes, int n_in,
                              void* d_out, int out_size, void* d_ws, size_t ws_size,
                              hipStream_t stream) {
  (void)in_sizes; (void)n_in; (void)out_size; (void)ws_size;
  const float* x    = (const float*)d_in[0];
  const float* Wq   = (const float*)d_in[1];
  const float* bq   = (const float*)d_in[2];
  const float* Wk   = (const float*)d_in[3];
  const float* bk   = (const float*)d_in[4];
  const float* Wv   = (const float*)d_in[5];
  const float* bv   = (const float*)d_in[6];
  const float* relh = (const float*)d_in[7];
  const float* relw = (const float*)d_in[8];
  const float* lng  = (const float*)d_in[9];
  const float* lnb  = (const float*)d_in[10];
  const float* caw1 = (const float*)d_in[11];
  const float* cab1 = (const float*)d_in[12];
  const float* caw2 = (const float*)d_in[13];
  const float* cab2 = (const float*)d_in[14];
  const float* saw  = (const float*)d_in[15];
  const float* sab  = (const float*)d_in[16];
  float* out = (float*)d_out;

  char* ws = (char*)d_ws;
  size_t off = 0;
  auto alloc = [&](size_t bytes) -> char* {
    char* p = ws + off;
    off = (off + bytes + 255) & ~(size_t)255;
    return p;
  };
  const size_t nBHND = (size_t)B_ * HEADS_ * N_ * D_;     // 2,097,152
  _Float16* qt  = (_Float16*)alloc(nBHND * 2);
  _Float16* kt  = (_Float16*)alloc(nBHND * 2);
  _Float16* vd  = (_Float16*)alloc(nBHND * 2);
  _Float16* pt  = (_Float16*)alloc((size_t)HEADS_ * N_ * D_ * 2);
  float* att    = (float*)alloc((size_t)B_ * C_ * N_ * 4);
  float* mu     = (float*)alloc(B_ * 4);
  float* rstd   = (float*)alloc(B_ * 4);
  float* avg    = (float*)alloc(B_ * C_ * 4);
  float* mx     = (float*)alloc(B_ * C_ * 4);
  float* ch     = (float*)alloc(B_ * C_ * 4);
  float* fm     = (float*)alloc((size_t)B_ * N_ * 4);
  float* fx     = (float*)alloc((size_t)B_ * N_ * 4);
  float* spat   = (float*)alloc((size_t)B_ * N_ * 4);

  pos_kernel     <<<16,   256, 0, stream>>>(relh, relw, pt);
  qkv_wmma_kernel<<<256,  256, 0, stream>>>(x, Wq, bq, Wk, bk, Wv, bv, qt, kt, vd);
  attn_kernel    <<<512,  256, 0, stream>>>(qt, kt, vd, pt, att);
  lnstat_kernel  <<<32,   256, 0, stream>>>(att, mu, rstd);
  pool_kernel    <<<2048, 256, 0, stream>>>(x, avg, mx);
  mlp_kernel     <<<32,   64,  0, stream>>>(avg, mx, caw1, cab1, caw2, cab2, ch);
  feat_kernel    <<<128,  256, 0, stream>>>(x, ch, fm, fx);
  sconv_kernel   <<<128,  256, 0, stream>>>(fm, fx, saw, sab, spat);
  final_kernel   <<<8192, 256, 0, stream>>>(att, x, mu, rstd, lng, lnb, ch, spat, out);
}
